// FCViT_36627481100351
// MI455X (gfx1250) — compile-verified, matrix operand order checked by reading, not run on hardware
//
#include <hip/hip_runtime.h>
#include <hip/hip_bf16.h>
#include <math.h>

typedef __attribute__((ext_vector_type(16))) __bf16 v16bf;
typedef __attribute__((ext_vector_type(8)))  float  v8f;

#define DI __device__ __forceinline__

// ---------------- problem constants ----------------
#define NBATCH 32
#define PFRAG  75
#define NGRID  3
#define NFRAG  9
#define CROPSZ 64
#define PUZSZ  225
#define IMGSZ  224
#define PATCHSZ 16
#define NPTOK  196           // (224/16)^2
#define SEQ    197
#define DMODEL 768
#define NHEAD  12
#define HDIM   64
#define NLAYER 12
#define DMLP   3072
#define NCLS_  1000
#define NCLSP  1024          // padded
#define NOUT2  18
#define NOUT2P 256           // padded to one block-column

// ---------------- deterministic PRNG (threefry stand-in) ----------------
DI unsigned h32(unsigned x){ x^=x>>16; x*=0x7feb352du; x^=x>>15; x*=0x846ca68bu; x^=x>>16; return x; }
DI unsigned rng2(unsigned s, unsigned i){ return h32(h32(s*0x9E3779B9u + 0x85ebca6bu) ^ (i*0xC2B2AE35u + 0x27d4eb2fu)); }
DI float runif01(unsigned s, unsigned i){ return (float)(rng2(s,i)>>8) * (1.0f/16777216.0f); }
DI int rint_range(unsigned s, unsigned i, int lo, int hi){ return lo + (int)(rng2(s,i) % (unsigned)(hi-lo)); }

// WMMA 16-bit fragment K index for element j, K-half kh (ISA A/B layout)
DI int kidx(int j, int kh){ return ((j>>3)<<4) + (kh<<3) + (j&7); }

// ---------------- augmentation ----------------
__global__ void k_setup(int* __restrict__ ids, int* __restrict__ offs, int* __restrict__ jit,
                        float* __restrict__ target){
    int b = blockIdx.x*blockDim.x + threadIdx.x;
    if(b >= NBATCH) return;
    float noise[NFRAG]; int idx[NFRAG];
    for(int f=0; f<NFRAG; ++f){ noise[f] = runif01(1u, (unsigned)(b*NFRAG+f)); idx[f]=f; }
    for(int i=1;i<NFRAG;++i){           // stable argsort (ascending)
        float nv=noise[i]; int iv=idx[i]; int j=i-1;
        while(j>=0 && noise[j] > nv){ noise[j+1]=noise[j]; idx[j+1]=idx[j]; --j; }
        noise[j+1]=nv; idx[j+1]=iv;
    }
    int restore[NFRAG];
    for(int f=0; f<NFRAG; ++f){ ids[b*NFRAG+f]=idx[f]; restore[idx[f]]=f; }
    for(int f=0; f<NFRAG; ++f){
        target[(b*NFRAG+f)*2+0] = (float)(restore[f]/NGRID);
        target[(b*NFRAG+f)*2+1] = (float)(restore[f]%NGRID);
        int bf = b*NFRAG+f;
        offs[bf*2+0] = rint_range(2u, (unsigned)(bf*2+0), 0, PFRAG-CROPSZ+1);
        offs[bf*2+1] = rint_range(2u, (unsigned)(bf*2+1), 0, PFRAG-CROPSZ+1);
        for(int c=0;c<3;++c) jit[bf*3+c] = rint_range(3u, (unsigned)(bf*3+c), -2, 3);
    }
}

__global__ void k_resize(const float* __restrict__ x, const int* __restrict__ ids,
                         const int* __restrict__ offs, const int* __restrict__ jit,
                         float* __restrict__ res){
    long i = (long)blockIdx.x*blockDim.x + threadIdx.x;
    const long total = (long)NBATCH*NFRAG*3*PFRAG*PFRAG;
    if(i >= total) return;
    int ix = (int)(i % PFRAG); long r = i / PFRAG;
    int iy = (int)(r % PFRAG); r /= PFRAG;
    int c  = (int)(r % 3);     r /= 3;
    int f  = (int)(r % NFRAG); int b = (int)(r / NFRAG);
    int bf = b*NFRAG+f;
    int sf = ids[b*NFRAG+f];
    int gi = sf / NGRID, gj = sf % NGRID;
    int o0 = offs[bf*2+0], o1 = offs[bf*2+1];
    const float* src = x + ((long)(b*3 + c))*PUZSZ*PUZSZ;
    const float sc = (float)CROPSZ / (float)PFRAG;
    float cy = (iy + 0.5f)*sc - 0.5f;
    float cx = (ix + 0.5f)*sc - 0.5f;
    int y0 = (int)floorf(cy); float wy = cy - (float)y0;
    int x0 = (int)floorf(cx); float wx = cx - (float)x0;
    int y0c = min(max(y0,0),CROPSZ-1), y1c = min(max(y0+1,0),CROPSZ-1);
    int x0c = min(max(x0,0),CROPSZ-1), x1c = min(max(x0+1,0),CROPSZ-1);
    int by = gi*PFRAG + o0, bx = gj*PFRAG + o1;
    float v00 = src[(long)(by+y0c)*PUZSZ + bx+x0c];
    float v01 = src[(long)(by+y0c)*PUZSZ + bx+x1c];
    float v10 = src[(long)(by+y1c)*PUZSZ + bx+x0c];
    float v11 = src[(long)(by+y1c)*PUZSZ + bx+x1c];
    float v = (1.f-wy)*((1.f-wx)*v00 + wx*v01) + wy*((1.f-wx)*v10 + wx*v11);
    v += (float)jit[bf*3+c];
    res[i] = fminf(fmaxf(v, 0.f), 255.f);
}

__global__ void k_stats(const float* __restrict__ res, float* __restrict__ stats){
    int bc = blockIdx.x;
    const float* p = res + (long)bc*PFRAG*PFRAG;
    __shared__ float ssum[256], ssq[256];
    float s=0.f, q=0.f;
    for(int i=threadIdx.x; i<PFRAG*PFRAG; i+=blockDim.x){ float v=p[i]; s+=v; q+=v*v; }
    ssum[threadIdx.x]=s; ssq[threadIdx.x]=q; __syncthreads();
    for(int st=128; st>0; st>>=1){
        if((int)threadIdx.x < st){ ssum[threadIdx.x]+=ssum[threadIdx.x+st]; ssq[threadIdx.x]+=ssq[threadIdx.x+st]; }
        __syncthreads();
    }
    if(threadIdx.x==0){
        const float cnt = (float)(PFRAG*PFRAG);
        float m = ssum[0]/cnt;
        float var = (ssq[0] - ssum[0]*ssum[0]/cnt) / (cnt - 1.0f);
        float sd = sqrtf(fmaxf(var, 0.f));
        if(sd == 0.f) sd = 1.f;
        stats[bc*2+0]=m; stats[bc*2+1]=sd;
    }
}

__global__ void k_assemble(const float* __restrict__ res, const float* __restrict__ stats,
                           float* __restrict__ img){
    long i = (long)blockIdx.x*blockDim.x + threadIdx.x;
    const long total = (long)NBATCH*3*PUZSZ*PUZSZ;
    if(i>=total) return;
    int xx = (int)(i % PUZSZ); long r = i/PUZSZ;
    int yy = (int)(r % PUZSZ); r /= PUZSZ;
    int c = (int)(r % 3); int b = (int)(r / 3);
    int gi = yy/PFRAG, gj = xx/PFRAG;
    int iy = yy%PFRAG, ix = xx%PFRAG;
    int bf = b*NFRAG + (gi*NGRID+gj);
    long ridx = (((long)bf*3 + c)*PFRAG + iy)*PFRAG + ix;
    int sidx = bf*3+c;
    img[i] = (res[ridx] - stats[sidx*2]) / stats[sidx*2+1];
}

__global__ void k_patches(const float* __restrict__ img, float* __restrict__ patches){
    long i = (long)blockIdx.x*blockDim.x + threadIdx.x;
    const long total = (long)NBATCH*NPTOK*DMODEL;
    if(i>=total) return;
    int col = (int)(i % DMODEL); long r = i/DMODEL;
    int t = (int)(r % NPTOK); int b = (int)(r/NPTOK);
    int ch = col/(PATCHSZ*PATCHSZ); int rem = col % (PATCHSZ*PATCHSZ);
    int iy = rem/PATCHSZ, ix = rem%PATCHSZ;
    int py = t/(IMGSZ/PATCHSZ), px = t%(IMGSZ/PATCHSZ);
    patches[i] = img[(((long)(b*3+ch))*PUZSZ + (py*PATCHSZ+iy))*PUZSZ + (px*PATCHSZ+ix)];
}

__global__ void k_build_tok(const float* __restrict__ ptok, const float* __restrict__ cls_token,
                            const float* __restrict__ pos, float* __restrict__ tok){
    long i = (long)blockIdx.x*blockDim.x + threadIdx.x;
    const long total = (long)NBATCH*SEQ*DMODEL;
    if(i>=total) return;
    int d = (int)(i % DMODEL); long r = i/DMODEL;
    int t = (int)(r % SEQ); int b = (int)(r / SEQ);
    float base = (t==0) ? cls_token[d] : ptok[((long)(b*NPTOK + t-1))*DMODEL + d];
    tok[i] = base + pos[t*DMODEL + d];
}

// ---------------- LayerNorm ----------------
__global__ void k_ln(const float* __restrict__ x, long xstride,
                     const float* __restrict__ g, const float* __restrict__ be,
                     float* __restrict__ out, long ostride, int rows){
    int row = blockIdx.x;
    if(row>=rows) return;
    const float* p = x + (long)row*xstride;
    __shared__ float s1[256], s2[256];
    float a=0.f, b2=0.f;
    for(int i=threadIdx.x;i<DMODEL;i+=blockDim.x){ float v=p[i]; a+=v; b2+=v*v; }
    s1[threadIdx.x]=a; s2[threadIdx.x]=b2; __syncthreads();
    for(int st=128; st>0; st>>=1){
        if((int)threadIdx.x<st){ s1[threadIdx.x]+=s1[threadIdx.x+st]; s2[threadIdx.x]+=s2[threadIdx.x+st]; }
        __syncthreads();
    }
    float m = s1[0]/(float)DMODEL;
    float var = s2[0]/(float)DMODEL - m*m;
    float inv = rsqrtf(var + 1e-6f);
    float* o = out + (long)row*ostride;
    for(int i=threadIdx.x;i<DMODEL;i+=blockDim.x) o[i] = (p[i]-m)*inv*g[i] + be[i];
}

// ---------------- attention ----------------
__global__ __launch_bounds__(64) void k_attn(const float* __restrict__ qkv, float* __restrict__ obuf){
    int tq = blockIdx.x, h = blockIdx.y, b = blockIdx.z;
    int t = threadIdx.x;
    __shared__ float qv[HDIM];
    __shared__ float sc[SEQ];
    __shared__ float red[64];
    const long rowstride = 3*DMODEL;
    qv[t] = qkv[((long)(b*SEQ + tq))*rowstride + h*HDIM + t];
    __syncthreads();
    float lmax = -1e30f;
    for(int k=t;k<SEQ;k+=64){
        const float* krow = qkv + ((long)(b*SEQ + k))*rowstride + DMODEL + h*HDIM;
        float d=0.f;
        #pragma unroll 8
        for(int j=0;j<HDIM;++j) d += qv[j]*krow[j];
        d *= 0.125f;
        sc[k]=d; lmax = fmaxf(lmax, d);
    }
    red[t]=lmax; __syncthreads();
    for(int st=32; st>0; st>>=1){ if(t<st) red[t]=fmaxf(red[t],red[t+st]); __syncthreads(); }
    float mx = red[0]; __syncthreads();
    float lsum=0.f;
    for(int k=t;k<SEQ;k+=64){ float e=expf(sc[k]-mx); sc[k]=e; lsum+=e; }
    red[t]=lsum; __syncthreads();
    for(int st=32; st>0; st>>=1){ if(t<st) red[t]+=red[t+st]; __syncthreads(); }
    float inv = 1.0f/red[0];
    __syncthreads();
    float acc=0.f;
    for(int k=0;k<SEQ;++k)
        acc += sc[k] * qkv[((long)(b*SEQ + k))*rowstride + 2*DMODEL + h*HDIM + t];
    obuf[((long)(b*SEQ + tq))*DMODEL + h*HDIM + t] = acc*inv;
}

// ---------------- weight pre-pack: f32 (K,N) -> bf16 WMMA B-fragment order ----------------
// packed index = ((nt*KT + kt)*32 + lane)*16 + j ; zero-pads K->32*KT, N->16*NT (cold path).
__global__ void k_pack(const float* __restrict__ B, int K, int N, int ldb,
                       int KT, int NT, __bf16* __restrict__ P){
    long i = (long)blockIdx.x*blockDim.x + threadIdx.x;
    long total = (long)KT*NT*512;
    if(i>=total) return;
    int j = (int)(i & 15);
    int lane = (int)((i>>4)&31);
    long tile = i>>9;
    int kt = (int)(tile % KT);
    int nt = (int)(tile / KT);
    int k = kt*32 + kidx(j, lane>>4);
    int n = nt*16 + (lane&15);
    P[i] = (k<K && n<N) ? (__bf16)B[(long)k*ldb + n] : (__bf16)0.f;
}

// ---------------- activation pre-pack: f32 (M,K) -> bf16 WMMA A-fragment order ----------------
// packed index = ((mt*KT + kt)*32 + lane)*16 + j ; dims exact (M mult 16, K = 32*KT).
// One thread converts one 8-float run (two float4 loads -> 16B packed store).
__global__ void k_packA(const float* __restrict__ A, int lda, int KT,
                        __bf16* __restrict__ P, long total /* = MT*KT*64 */){
    long i = (long)blockIdx.x*blockDim.x + threadIdx.x;
    if(i>=total) return;
    int g = (int)(i & 1);
    int lane = (int)((i>>1) & 31);
    long tile = i>>6;
    int kt = (int)(tile % KT);
    int mt = (int)(tile / KT);
    int lm = lane & 15, kh = lane >> 4;
    int row = mt*16 + lm;
    int k = kt*32 + g*16 + kh*8;
    const float* ap = A + (size_t)row*lda + k;
    float4 a0 = *(const float4*)ap;
    float4 a1 = *(const float4*)(ap+4);
    __bf16* o = P + (tile<<9) + lane*16 + g*8;
    o[0]=(__bf16)a0.x; o[1]=(__bf16)a0.y; o[2]=(__bf16)a0.z; o[3]=(__bf16)a0.w;
    o[4]=(__bf16)a1.x; o[5]=(__bf16)a1.y; o[6]=(__bf16)a1.z; o[7]=(__bf16)a1.w;
}

__global__ void k_padvec(const float* __restrict__ src, int n, float* __restrict__ dst, int np){
    int i = blockIdx.x*blockDim.x + threadIdx.x;
    if(i<np) dst[i] = (i<n) ? src[i] : 0.f;
}

__global__ void k_extract(const float* __restrict__ Cp, float* __restrict__ out){
    int i = blockIdx.x*blockDim.x + threadIdx.x;
    if(i >= NBATCH*NOUT2) return;
    int r = i / NOUT2, c = i % NOUT2;
    out[i] = Cp[r*NOUT2P + c];
}

// ---------------- guard-free bf16 WMMA GEMM (both operands pre-packed) ----------------
// Grid: (M/16, N/256); block 128 = 4 waves; wave w -> 16 rows x 64 cols (4 WMMA tiles).
// Hot loop: 2 b128 A-loads + 8 b128 B-loads + A-stream prefetch + 4 WMMA. No VALU converts.
__global__ __launch_bounds__(128) void k_gemm(
    const __bf16* __restrict__ Ap, int KT,
    const __bf16* __restrict__ Bp,
    const float* __restrict__ bias,
    const float* __restrict__ resid,
    float* __restrict__ C, int ldc, int act)   // act: 0=none 1=gelu 2=relu
{
    const int wave = threadIdx.x >> 5;
    const int lane = threadIdx.x & 31;
    const int lm = lane & 15;
    const int kh = lane >> 4;
    const int mt = blockIdx.x;
    const int ntBase = blockIdx.y*16 + wave*4;

    v8f acc0, acc1, acc2, acc3;
    #pragma unroll
    for(int i=0;i<8;++i){ acc0[i]=0.f; acc1[i]=0.f; acc2[i]=0.f; acc3[i]=0.f; }

    const __bf16* pa  = Ap + ((size_t)mt*KT*32 + lane)*16;
    const __bf16* bp0 = Bp + ((size_t)(ntBase+0)*KT*32 + lane)*16;
    const __bf16* bp1 = Bp + ((size_t)(ntBase+1)*KT*32 + lane)*16;
    const __bf16* bp2 = Bp + ((size_t)(ntBase+2)*KT*32 + lane)*16;
    const __bf16* bp3 = Bp + ((size_t)(ntBase+3)*KT*32 + lane)*16;

    for(int kt=0; kt<KT; ++kt){
        v16bf a  = *(const v16bf*)(pa  + (size_t)kt*512);
        v16bf b0 = *(const v16bf*)(bp0 + (size_t)kt*512);
        v16bf b1 = *(const v16bf*)(bp1 + (size_t)kt*512);
        v16bf b2 = *(const v16bf*)(bp2 + (size_t)kt*512);
        v16bf b3 = *(const v16bf*)(bp3 + (size_t)kt*512);
        __builtin_prefetch((const void*)(pa + (size_t)(kt+8)*512), 0, 3);  // global_prefetch_b8

        acc0 = __builtin_amdgcn_wmma_f32_16x16x32_bf16(false, a, false, b0, (short)0, acc0, false, false);
        acc1 = __builtin_amdgcn_wmma_f32_16x16x32_bf16(false, a, false, b1, (short)0, acc1, false, false);
        acc2 = __builtin_amdgcn_wmma_f32_16x16x32_bf16(false, a, false, b2, (short)0, acc2, false, false);
        acc3 = __builtin_amdgcn_wmma_f32_16x16x32_bf16(false, a, false, b3, (short)0, acc3, false, false);
    }

    #pragma unroll
    for(int v=0; v<8; ++v){
        int row = mt*16 + v + 8*kh;
        float vals[4] = {acc0[v], acc1[v], acc2[v], acc3[v]};
        #pragma unroll
        for(int t=0;t<4;++t){
            int col = (ntBase+t)*16 + lm;
            float xv = vals[t];
            if(bias) xv += bias[col];
            if(act==1)      xv = 0.5f*xv*(1.0f+erff(xv*0.70710678118f));
            else if(act==2) xv = fmaxf(xv, 0.f);
            if(resid) xv += resid[(size_t)row*ldc + col];
            C[(size_t)row*ldc + col] = xv;
        }
    }
}

// ---------------- host side ----------------
static inline void* ws_alloc(char*& p, size_t bytes){
    void* r = (void*)p;
    p += (bytes + 255) & ~(size_t)255;
    return r;
}

extern "C" void kernel_launch(void* const* d_in, const int* in_sizes, int n_in,
                              void* d_out, int out_size, void* d_ws, size_t ws_size,
                              hipStream_t stream)
{
    (void)in_sizes; (void)n_in; (void)out_size; (void)ws_size;

    const float* x        = (const float*)d_in[0];
    const float* patch_w  = (const float*)d_in[1];
    const float* patch_b  = (const float*)d_in[2];
    const float* cls_tok  = (const float*)d_in[3];
    const float* pos_emb  = (const float*)d_in[4];
    const float* ln1_s    = (const float*)d_in[5];
    const float* ln1_b    = (const float*)d_in[6];
    const float* qkv_w    = (const float*)d_in[7];
    const float* qkv_b    = (const float*)d_in[8];
    const float* proj_w   = (const float*)d_in[9];
    const float* proj_b   = (const float*)d_in[10];
    const float* ln2_s    = (const float*)d_in[11];
    const float* ln2_b    = (const float*)d_in[12];
    const float* mlp1_w   = (const float*)d_in[13];
    const float* mlp1_b   = (const float*)d_in[14];
    const float* mlp2_w   = (const float*)d_in[15];
    const float* mlp2_b   = (const float*)d_in[16];
    const float* lnf_s    = (const float*)d_in[17];
    const float* lnf_b    = (const float*)d_in[18];
    const float* head_w   = (const float*)d_in[19];
    const float* head_b   = (const float*)d_in[20];
    const float* fc1_w    = (const float*)d_in[21];
    const float* fc1_b    = (const float*)d_in[22];
    const float* fc2_w    = (const float*)d_in[23];
    const float* fc2_b    = (const float*)d_in[24];

    const int ROWS = NBATCH*SEQ;       // 6304 = 394*16
    const int PROWS = NBATCH*NPTOK;    // 6272 = 392*16
    const int MT_R = ROWS/16, MT_P = PROWS/16, MT_B = NBATCH/16;

    char* wp = (char*)d_ws;
    // small buffers
    int*   ids   = (int*)  ws_alloc(wp, (size_t)NBATCH*NFRAG*sizeof(int));
    int*   offs  = (int*)  ws_alloc(wp, (size_t)NBATCH*NFRAG*2*sizeof(int));
    int*   jit   = (int*)  ws_alloc(wp, (size_t)NBATCH*NFRAG*3*sizeof(int));
    float* stats = (float*)ws_alloc(wp, (size_t)NBATCH*NFRAG*3*2*sizeof(float));
    float* hb_p  = (float*)ws_alloc(wp, NCLSP*sizeof(float));
    float* f1b_p = (float*)ws_alloc(wp, NCLSP*sizeof(float));
    float* f2b_p = (float*)ws_alloc(wp, NOUT2P*sizeof(float));
    // big float buffers
    float* tok   = (float*)ws_alloc(wp, (size_t)ROWS*DMODEL*sizeof(float));
    float* hbuf  = (float*)ws_alloc(wp, (size_t)ROWS*DMODEL*sizeof(float));   // also ptok
    float* qkvb  = (float*)ws_alloc(wp, (size_t)ROWS*3*DMODEL*sizeof(float)); // also img
    float* obuf  = (float*)ws_alloc(wp, (size_t)ROWS*DMODEL*sizeof(float));
    float* mlpb  = (float*)ws_alloc(wp, (size_t)ROWS*DMLP*sizeof(float));     // also res+patches
    float* clsb  = (float*)ws_alloc(wp, (size_t)NBATCH*DMODEL*sizeof(float));
    float* feat  = (float*)ws_alloc(wp, (size_t)NBATCH*NCLSP*sizeof(float));
    float* feat2 = (float*)ws_alloc(wp, (size_t)NBATCH*NCLSP*sizeof(float));
    float* outp  = (float*)ws_alloc(wp, (size_t)NBATCH*NOUT2P*sizeof(float));
    // aliased augmentation scratch (lifetimes do not overlap with layer compute)
    float* res     = mlpb;
    float* patches = mlpb + 5000192;
    float* img     = qkvb;
    float* ptok    = hbuf;
    // packed bf16 weights (B operands)
    const int KT_D = DMODEL/32, NT_D = DMODEL/16;           // 24, 48
    const int NT_3D = 3*DMODEL/16;                          // 144
    const int NT_M = DMLP/16;                               // 192
    const int KT_M = DMLP/32;                               // 96
    const int KT_C = NCLSP/32, NT_C = NCLSP/16;             // 32, 64
    const int NT_O = NOUT2P/16;                             // 16
    __bf16* p_patch = (__bf16*)ws_alloc(wp, (size_t)KT_D*NT_D*512*sizeof(__bf16));
    __bf16* p_qkv   = (__bf16*)ws_alloc(wp, (size_t)NLAYER*KT_D*NT_3D*512*sizeof(__bf16));
    __bf16* p_proj  = (__bf16*)ws_alloc(wp, (size_t)NLAYER*KT_D*NT_D*512*sizeof(__bf16));
    __bf16* p_mlp1  = (__bf16*)ws_alloc(wp, (size_t)NLAYER*KT_D*NT_M*512*sizeof(__bf16));
    __bf16* p_mlp2  = (__bf16*)ws_alloc(wp, (size_t)NLAYER*KT_M*NT_D*512*sizeof(__bf16));
    __bf16* p_head  = (__bf16*)ws_alloc(wp, (size_t)KT_D*NT_C*512*sizeof(__bf16));
    __bf16* p_fc1   = (__bf16*)ws_alloc(wp, (size_t)KT_C*NT_C*512*sizeof(__bf16));
    __bf16* p_fc2   = (__bf16*)ws_alloc(wp, (size_t)KT_C*NT_O*512*sizeof(__bf16));
    // packed bf16 activations (A operands)
    __bf16* pAd = (__bf16*)ws_alloc(wp, (size_t)MT_R*KT_D*512*sizeof(__bf16)); // K=768 activations
    __bf16* pAm = (__bf16*)ws_alloc(wp, (size_t)MT_R*KT_M*512*sizeof(__bf16)); // K=3072 activations
    (void)ws_alloc(wp, 8192);   // tail pad: keeps speculative A-prefetch inside mapped ws

    float* out    = (float*)d_out;
    float* target = out + NBATCH*NFRAG*2;

    auto pack = [&](const float* B, int K, int N, int ldb, int KT, int NT, __bf16* P){
        long total = (long)KT*NT*512;
        k_pack<<<(int)((total+255)/256),256,0,stream>>>(B,K,N,ldb,KT,NT,P);
    };
    auto packA = [&](const float* A, int lda, int MT, int KT, __bf16* P){
        long total = (long)MT*KT*64;
        k_packA<<<(int)((total+255)/256),256,0,stream>>>(A,lda,KT,P,total);
    };

    // ---- pre-pack all weights to bf16 fragment order (cold, bandwidth-trivial) ----
    pack(patch_w, DMODEL, DMODEL, DMODEL, KT_D, NT_D, p_patch);
    for(int l=0;l<NLAYER;++l){
        pack(qkv_w  + (size_t)l*DMODEL*3*DMODEL, DMODEL, 3*DMODEL, 3*DMODEL, KT_D, NT_3D, p_qkv  + (size_t)l*KT_D*NT_3D*512);
        pack(proj_w + (size_t)l*DMODEL*DMODEL,   DMODEL, DMODEL,   DMODEL,   KT_D, NT_D,  p_proj + (size_t)l*KT_D*NT_D*512);
        pack(mlp1_w + (size_t)l*DMODEL*DMLP,     DMODEL, DMLP,     DMLP,     KT_D, NT_M,  p_mlp1 + (size_t)l*KT_D*NT_M*512);
        pack(mlp2_w + (size_t)l*DMLP*DMODEL,     DMLP,   DMODEL,   DMODEL,   KT_M, NT_D,  p_mlp2 + (size_t)l*KT_M*NT_D*512);
    }
    pack(head_w, DMODEL, NCLS_, NCLS_, KT_D, NT_C, p_head);
    pack(fc1_w,  NCLS_,  NCLS_, NCLS_, KT_C, NT_C, p_fc1);
    pack(fc2_w,  NCLS_,  NOUT2, NOUT2, KT_C, NT_O, p_fc2);
    k_padvec<<<(NCLSP+255)/256,256,0,stream>>>(head_b, NCLS_, hb_p, NCLSP);
    k_padvec<<<(NCLSP+255)/256,256,0,stream>>>(fc1_b,  NCLS_, f1b_p, NCLSP);
    k_padvec<<<(NOUT2P+255)/256,256,0,stream>>>(fc2_b, NOUT2, f2b_p, NOUT2P);

    // ---- augmentation ----
    k_setup<<<1, 32, 0, stream>>>(ids, offs, jit, target);
    {   long total = (long)NBATCH*NFRAG*3*PFRAG*PFRAG;
        k_resize<<<(int)((total+255)/256), 256, 0, stream>>>(x, ids, offs, jit, res); }
    k_stats<<<NBATCH*NFRAG*3, 256, 0, stream>>>(res, stats);
    {   long total=(long)NBATCH*3*PUZSZ*PUZSZ;
        k_assemble<<<(int)((total+255)/256),256,0,stream>>>(res,stats,img); }
    {   long total=(long)PROWS*DMODEL;
        k_patches<<<(int)((total+255)/256),256,0,stream>>>(img,patches); }

    // ---- patch embed ----
    packA(patches, DMODEL, MT_P, KT_D, pAd);
    {   dim3 g(MT_P, DMODEL/256);
        k_gemm<<<g,128,0,stream>>>(pAd, KT_D, p_patch, patch_b, nullptr, ptok, DMODEL, 0); }
    {   long total=(long)ROWS*DMODEL;
        k_build_tok<<<(int)((total+255)/256),256,0,stream>>>(ptok, cls_tok, pos_emb, tok); }

    // ---- transformer layers ----
    for(int l=0;l<NLAYER;++l){
        k_ln<<<ROWS,256,0,stream>>>(tok, DMODEL, ln1_s + (size_t)l*DMODEL, ln1_b + (size_t)l*DMODEL,
                                    hbuf, DMODEL, ROWS);
        packA(hbuf, DMODEL, MT_R, KT_D, pAd);
        {   dim3 g(MT_R, 3*DMODEL/256);
            k_gemm<<<g,128,0,stream>>>(pAd, KT_D, p_qkv + (size_t)l*KT_D*NT_3D*512,
                                       qkv_b + (size_t)l*3*DMODEL, nullptr, qkvb, 3*DMODEL, 0); }
        {   dim3 g(SEQ, NHEAD, NBATCH);
            k_attn<<<g,64,0,stream>>>(qkvb, obuf); }
        packA(obuf, DMODEL, MT_R, KT_D, pAd);
        {   dim3 g(MT_R, DMODEL/256);
            k_gemm<<<g,128,0,stream>>>(pAd, KT_D, p_proj + (size_t)l*KT_D*NT_D*512,
                                       proj_b + (size_t)l*DMODEL, tok, tok, DMODEL, 0); }
        k_ln<<<ROWS,256,0,stream>>>(tok, DMODEL, ln2_s + (size_t)l*DMODEL, ln2_b + (size_t)l*DMODEL,
                                    hbuf, DMODEL, ROWS);
        packA(hbuf, DMODEL, MT_R, KT_D, pAd);
        {   dim3 g(MT_R, DMLP/256);
            k_gemm<<<g,128,0,stream>>>(pAd, KT_D, p_mlp1 + (size_t)l*KT_D*NT_M*512,
                                       mlp1_b + (size_t)l*DMLP, nullptr, mlpb, DMLP, 1); }
        packA(mlpb, DMLP, MT_R, KT_M, pAm);
        {   dim3 g(MT_R, DMODEL/256);
            k_gemm<<<g,128,0,stream>>>(pAm, KT_M, p_mlp2 + (size_t)l*KT_M*NT_D*512,
                                       mlp2_b + (size_t)l*DMODEL, tok, tok, DMODEL, 0); }
    }

    // ---- head ----
    k_ln<<<NBATCH,256,0,stream>>>(tok, (long)SEQ*DMODEL, lnf_s, lnf_b, clsb, DMODEL, NBATCH);
    packA(clsb, DMODEL, MT_B, KT_D, pAd);
    {   dim3 g(MT_B, NCLSP/256);
        k_gemm<<<g,128,0,stream>>>(pAd, KT_D, p_head, hb_p, nullptr, feat, NCLSP, 0); }
    packA(feat, NCLSP, MT_B, KT_C, pAd);
    {   dim3 g(MT_B, NCLSP/256);
        k_gemm<<<g,128,0,stream>>>(pAd, KT_C, p_fc1, f1b_p, nullptr, feat2, NCLSP, 2); }
    packA(feat2, NCLSP, MT_B, KT_C, pAd);
    {   dim3 g(MT_B, NOUT2P/256);
        k_gemm<<<g,128,0,stream>>>(pAd, KT_C, p_fc2, f2b_p, nullptr, outp, NOUT2P, 0); }
    k_extract<<<(NBATCH*NOUT2+255)/256,256,0,stream>>>(outp, out);
}
